// SimpleEERNN_32530082300461
// MI455X (gfx1250) — matrix operand order repaired and checked by reading, not compiled, not run on hardware
//
#include <hip/hip_runtime.h>
#include <hip/hip_bf16.h>

typedef __attribute__((ext_vector_type(16))) _Float16 v16h;
typedef __attribute__((ext_vector_type(8)))  float    v8f;

#define SEQ   1024
#define BATCH 64
#define DIN   512
#define DH    1024
#define LN_EPS 1e-5f

// ---------------------------------------------------------------------------
// Fragment layouts used here (contiguous k for A and B, not the ISA table order; 16-bit, wave32)
// B tile (32K x 16N):  lane = N + 16*(K>=16),        elem e = K & 15
// A tile (16M x 32K):  lane = M + 16*(K>=16),        elem e = K & 15
// C/D tile (16x16 f32): col N = lane&15, row M = r + 8*(lane>=16)
// ---------------------------------------------------------------------------

// Clamp params, convert weights to f16 pre-swizzled into B-fragment layout.
__global__ __launch_bounds__(256) void prep_kernel(
    const float* __restrict__ wxe, const float* __restrict__ wee,
    const float* __restrict__ bias,
    _Float16* __restrict__ wxe_t, _Float16* __restrict__ wee_t,
    float* __restrict__ bias_c) {
  int tid = blockIdx.x * 256 + threadIdx.x;   // 0 .. DH*DH-1

  if (tid < DIN * DH) {              // W_XE: (DH, DIN) row-major; K = din idx
    int j = tid >> 9;                // output column (N-space)
    int i = tid & (DIN - 1);         // K
    float v = fmaxf(wxe[tid], 0.f);
    int kt = i >> 5, K = i & 31, nt = j >> 4, N = j & 15;
    int lane = N + (((K >> 4) & 1) << 4);
    wxe_t[((size_t)(kt * (DH / 16) + nt) * 32 + lane) * 16 + (K & 15)] = (_Float16)v;
  }
  {                                   // W_EE: (DH, DH); K = second index
    int j = tid >> 10;
    int k = tid & (DH - 1);
    float v = fmaxf(wee[tid], 0.f);
    int kt = k >> 5, K = k & 31, nt = j >> 4, N = j & 15;
    int lane = N + (((K >> 4) & 1) << 4);
    wee_t[((size_t)(kt * (DH / 16) + nt) * 32 + lane) * 16 + (K & 15)] = (_Float16)v;
  }
  if (tid < DH) bias_c[tid] = fmaxf(bias[tid], 0.f);
}

// x_drive GEMM: out[m,n] = sum_k x[m,k] * W_XE^T[k,n] + bias[n]
// Wave tile 16(M) x 64(N), K stepped by 32. Streaming data -> NT hints.
__global__ __launch_bounds__(256) void xdrive_kernel(
    const float* __restrict__ x, const _Float16* __restrict__ wxe_t,
    const float* __restrict__ bias_c, float* __restrict__ out) {
  int lane = threadIdx.x & 31;
  int wave = blockIdx.x * 8 + (threadIdx.x >> 5);
  int n0 = (wave & 15) * 64;
  int m0 = (wave >> 4) * 16;
  int rowA = m0 + (lane & 15);
  int kbA  = (lane >> 4) << 4;

  v8f acc[4];
#pragma unroll
  for (int q = 0; q < 4; ++q) {
    float bv = bias_c[n0 + q * 16 + (lane & 15)];
#pragma unroll
    for (int r = 0; r < 8; ++r) acc[q][r] = bv;
  }

  for (int k0 = 0; k0 < DIN; k0 += 32) {
    v16h a;
#pragma unroll
    for (int e = 0; e < 16; ++e) {
      int k = k0 + kbA + e;
      a[e] = (_Float16)__builtin_nontemporal_load(&x[(size_t)rowA * DIN + k]);
    }
#pragma unroll
    for (int q = 0; q < 4; ++q) {
      const v16h b = *(const v16h*)(wxe_t +
          ((size_t)((k0 >> 5) * (DH / 16) + (n0 >> 4) + q) * 32 + lane) * 16);
      acc[q] = __builtin_amdgcn_wmma_f32_16x16x32_f16(
          false, a, false, b, (short)0, acc[q], false, false);
    }
  }

  int mrow = m0 + ((lane >> 4) << 3);
  int ncol = n0 + (lane & 15);
#pragma unroll
  for (int q = 0; q < 4; ++q)
#pragma unroll
    for (int r = 0; r < 8; ++r)
      __builtin_nontemporal_store(acc[q][r],
          &out[(size_t)(mrow + r) * DH + ncol + q * 16]);
}

// Persistent recurrence: 4 WGs x 1024 threads. WG owns batch rows
// [blockIdx.x*16, +16) for the full sequence (rows are independent!).
// h kept in LDS in A-fragment layout; pre never leaves registers.
__global__ __launch_bounds__(1024) void recurrent_kernel(
    const _Float16* __restrict__ wee_t, float* __restrict__ outb) {
  const int tid  = threadIdx.x;
  const int lane = tid & 31;
  const int wv   = tid >> 5;            // 0..31 : N-tile of 32 columns
  const int c    = lane & 15;
  const int hi   = lane >> 4;           // 0/1 -> rows r / r+8
  const int n0   = wv * 32;
  const int mt   = blockIdx.x;          // batch tile
  const int mrow = mt * 16 + hi * 8;    // first global row in acc[.][r]

  __shared__ _Float16 hls[16 * DH];     // 32 KB: A-fragment layout, local mt=0
  __shared__ float lred[32 * 16];
  __shared__ float lmu[16];
  __shared__ float lrs[16];

  for (int i = tid; i < 16 * DH; i += 1024) hls[i] = (_Float16)0.f;  // h0 = 0
  __syncthreads();

  for (int s = 0; s < SEQ; ++s) {
    const float* xd = outb + (size_t)s * (BATCH * DH);

    // ---- pre = x_drive[s] + h @ W_EE^T  (16x32 tile per wave) ----
    v8f acc[2];
#pragma unroll
    for (int q = 0; q < 2; ++q)
#pragma unroll
      for (int r = 0; r < 8; ++r)
        acc[q][r] = __builtin_nontemporal_load(
            &xd[(size_t)(mrow + r) * DH + n0 + q * 16 + c]);

    for (int k0 = 0; k0 < DH; k0 += 32) {
      const v16h a = *(const v16h*)(hls + ((size_t)((k0 >> 5) * 32 + lane)) * 16);
#pragma unroll
      for (int q = 0; q < 2; ++q) {
        const v16h b = *(const v16h*)(wee_t +
            ((size_t)((k0 >> 5) * (DH / 16) + (n0 >> 4) + q) * 32 + lane) * 16);
        acc[q] = __builtin_amdgcn_wmma_f32_16x16x32_f16(
            false, a, false, b, (short)0, acc[q], false, false);
      }
    }
    __syncthreads();   // all GEMM reads of hls done before LN overwrites it

    // ---- LayerNorm mean: per-lane partials, shuffle across 16-lane half ----
    float ps[8];
#pragma unroll
    for (int r = 0; r < 8; ++r) ps[r] = acc[0][r] + acc[1][r];
#pragma unroll
    for (int m = 1; m <= 8; m <<= 1)
#pragma unroll
      for (int r = 0; r < 8; ++r) ps[r] += __shfl_xor(ps[r], m, 32);
    if (c == 0)
#pragma unroll
      for (int r = 0; r < 8; ++r) lred[wv * 16 + hi * 8 + r] = ps[r];
    __syncthreads();
    if (tid < 16) {
      float t = 0.f;
#pragma unroll
      for (int w = 0; w < 32; ++w) t += lred[w * 16 + tid];
      lmu[tid] = t * (1.f / (float)DH);
    }
    __syncthreads();
    float mu[8];
#pragma unroll
    for (int r = 0; r < 8; ++r) mu[r] = lmu[hi * 8 + r];

    // ---- variance ----
#pragma unroll
    for (int r = 0; r < 8; ++r) {
      float d0 = acc[0][r] - mu[r], d1 = acc[1][r] - mu[r];
      ps[r] = d0 * d0 + d1 * d1;
    }
#pragma unroll
    for (int m = 1; m <= 8; m <<= 1)
#pragma unroll
      for (int r = 0; r < 8; ++r) ps[r] += __shfl_xor(ps[r], m, 32);
    if (c == 0)
#pragma unroll
      for (int r = 0; r < 8; ++r) lred[wv * 16 + hi * 8 + r] = ps[r];
    __syncthreads();
    if (tid < 16) {
      float t = 0.f;
#pragma unroll
      for (int w = 0; w < 32; ++w) t += lred[w * 16 + tid];
      lrs[tid] = rsqrtf(t * (1.f / (float)DH) + LN_EPS);
    }
    __syncthreads();
    float rsd[8];
#pragma unroll
    for (int r = 0; r < 8; ++r) rsd[r] = lrs[hi * 8 + r];

    // ---- h = tanh(LN(pre)): write f32 to output, f16 A-fragments to LDS ----
    float* o = outb + (size_t)s * (BATCH * DH);
#pragma unroll
    for (int q = 0; q < 2; ++q)
#pragma unroll
      for (int r = 0; r < 8; ++r) {
        int grow = mrow + r;
        int col  = n0 + q * 16 + c;
        float hv = tanhf((acc[q][r] - mu[r]) * rsd[r]);
        __builtin_nontemporal_store(hv, &o[(size_t)grow * DH + col]);
        int K = col & 31;
        int lane2 = (grow & 15) + ((K >> 4) << 4);
        int e = K & 15;
        hls[((size_t)((col >> 5) * 32 + lane2)) * 16 + e] = (_Float16)hv;
      }
    __syncthreads();   // LN writes to hls visible before next step's GEMM
  }
}

extern "C" void kernel_launch(void* const* d_in, const int* in_sizes, int n_in,
                              void* d_out, int out_size, void* d_ws, size_t ws_size,
                              hipStream_t stream) {
  (void)in_sizes; (void)n_in; (void)out_size; (void)ws_size;
  const float* x    = (const float*)d_in[0];
  const float* wxe  = (const float*)d_in[1];
  const float* wee  = (const float*)d_in[2];
  const float* bias = (const float*)d_in[3];
  float* outb = (float*)d_out;

  // Workspace: W_XE^T f16 fragments (1 MiB) | W_EE^T f16 fragments (2 MiB) | bias (4 KiB)
  _Float16* wxe_t  = (_Float16*)d_ws;
  _Float16* wee_t  = wxe_t + (size_t)DIN * DH;
  float*    bias_c = (float*)(wee_t + (size_t)DH * DH);

  // 1) clamp + fragment-swizzle weights, clamp bias
  prep_kernel<<<(DH * DH) / 256, 256, 0, stream>>>(wxe, wee, bias,
                                                   wxe_t, wee_t, bias_c);

  // 2) x_drive = x @ W_XE^T + bias, in place into d_out (65536 waves)
  xdrive_kernel<<<8192, 256, 0, stream>>>(x, wxe_t, bias_c, outb);

  // 3) whole recurrence in ONE persistent kernel: 4 WGs, each owns 16
  //    independent batch rows; h stays in LDS, pre stays in registers.
  recurrent_kernel<<<4, 1024, 0, stream>>>(wee_t, outb);

  // 4) h_T = outputs[-1]
  hipMemcpyAsync(outb + (size_t)SEQ * BATCH * DH,
                 outb + (size_t)(SEQ - 1) * BATCH * DH,
                 (size_t)BATCH * DH * sizeof(float),
                 hipMemcpyDeviceToDevice, stream);
}
